// ReparameterizedHMM_70334384439669
// MI455X (gfx1250) — 8-bit matrix operand order repaired and checked by reading, compiled, not run on hardware
//
#include <hip/hip_runtime.h>
#include <hip/hip_bf16.h>

#define S 512
#define V 32000
#define B 16
#define T 128

#define LN2    0.6931471805599453f
#define LOG2E  1.4426950408889634f

typedef __attribute__((ext_vector_type(16))) int   v16i;
typedef __attribute__((ext_vector_type(8)))  float v8f;

__device__ __forceinline__ unsigned char f32_to_fp8(float x) {
  if (!(x >= 9.765625e-4f)) return 0;
  unsigned u = __float_as_uint(x);
  int e = (int)((u >> 23) & 0xFF) - 127;
  if (e < -6) {
    int m = (int)(x * 512.0f + 0.5f);
    if (m > 7) return 0x08;
    return (unsigned char)m;
  }
  unsigned mant = (u >> 20) & 0x7;
  mant += (u >> 19) & 1;
  unsigned exp8 = (unsigned)(e + 7);
  if (mant > 7) { mant = 0; exp8 += 1; }
  if (exp8 > 15) { exp8 = 15; mant = 6; }
  return (unsigned char)((exp8 << 3) | (mant & 7));
}

__global__ void hmm_zem(const float* __restrict__ em, float* __restrict__ Zem) {
  __shared__ float sm[8], ss[8];
  const int row = blockIdx.x, tid = threadIdx.x;
  const float* r = em + (long)row * V;
  float m = -__builtin_inff(), s = 0.f;
  for (int c = tid; c < V; c += 256) {
    float x = r[c];
    if (x > m) { s = s * __expf(m - x) + 1.f; m = x; }
    else       { s += __expf(x - m); }
  }
  #pragma unroll
  for (int off = 16; off; off >>= 1) {
    float m2 = __shfl_xor(m, off);
    float s2 = __shfl_xor(s, off);
    float M  = fmaxf(m, m2);
    s = s * __expf(m - M) + s2 * __expf(m2 - M);
    m = M;
  }
  const int wave = tid >> 5, lane = tid & 31;
  if (lane == 0) { sm[wave] = m; ss[wave] = s; }
  __syncthreads();
  if (tid == 0) {
    float M = sm[0], Sa = ss[0];
    for (int w = 1; w < 8; ++w) {
      float MM = fmaxf(M, sm[w]);
      Sa = Sa * __expf(M - MM) + ss[w] * __expf(sm[w] - MM);
      M = MM;
    }
    Zem[row] = M + logf(Sa);
  }
}

__global__ void hmm_ztm(const float* __restrict__ tm, float* __restrict__ Ztm) {
  __shared__ float buf[256];
  const int row = blockIdx.x, tid = threadIdx.x;
  float x0 = tm[row * S + tid];
  float x1 = tm[row * S + tid + 256];
  buf[tid] = fmaxf(x0, x1);
  __syncthreads();
  for (int off = 128; off; off >>= 1) {
    if (tid < off) buf[tid] = fmaxf(buf[tid], buf[tid + off]);
    __syncthreads();
  }
  float M = buf[0];
  __syncthreads();
  buf[tid] = __expf(x0 - M) + __expf(x1 - M);
  __syncthreads();
  for (int off = 128; off; off >>= 1) {
    if (tid < off) buf[tid] += buf[tid + off];
    __syncthreads();
  }
  if (tid == 0) Ztm[row] = M + logf(buf[0]);
}

__global__ void hmm_tcol(const float* __restrict__ tm, const float* __restrict__ Ztm,
                         unsigned char* __restrict__ Ttr, float* __restrict__ kcol) {
  __shared__ float buf[256];
  const int c = blockIdx.x, tid = threadIdx.x;
  float p0 = __expf(tm[tid * S + c]         - Ztm[tid]);
  float p1 = __expf(tm[(tid + 256) * S + c] - Ztm[tid + 256]);
  buf[tid] = fmaxf(p0, p1);
  __syncthreads();
  for (int off = 128; off; off >>= 1) {
    if (tid < off) buf[tid] = fmaxf(buf[tid], buf[tid + off]);
    __syncthreads();
  }
  float M = buf[0];
  int   e = (int)((__float_as_uint(M) >> 23) & 0xFF) - 126;
  int   k = 8 - e;
  float scale = ldexpf(1.0f, k);
  Ttr[c * S + tid]       = f32_to_fp8(p0 * scale);
  Ttr[c * S + tid + 256] = f32_to_fp8(p1 * scale);
  if (tid == 0) kcol[c] = (float)k;
}

__global__ void hmm_gather(const int* __restrict__ ids, const float* __restrict__ em,
                           const float* __restrict__ Zem, float* __restrict__ g) {
  const int tok = blockIdx.x;
  const int s   = threadIdx.x;
  const int id  = ids[tok];
  g[(long)tok * S + s] = em[(long)s * V + id] - Zem[s];
}

__global__ void __launch_bounds__(512, 1)
hmm_scan(const float* __restrict__ p, const float* __restrict__ Zem,
         const unsigned char* __restrict__ TtrG, const float* __restrict__ kcolG,
         const float* __restrict__ g, float* __restrict__ out) {
  extern __shared__ __align__(16) char smem[];
  float*         alpha = (float*)smem;
  float*         mrow  = (float*)(smem + 32768);
  float*         llrow = (float*)(smem + 32768 + 64);
  float*         zpS   = (float*)(smem + 32768 + 128);
  float*         kcolS = (float*)(smem + 33024);
  unsigned char* W8    = (unsigned char*)(smem + 35072);
  unsigned char* T8    = (unsigned char*)(smem + 43264);

  const int tid    = threadIdx.x;
  const int wave   = tid >> 5;
  const int lane   = tid & 31;
  const int laneHi = (lane >> 4) & 1;
  const int ln     = lane & 15;

  {
    uint4*       dst = (uint4*)T8;
    const uint4* src = (const uint4*)TtrG;
    #pragma unroll
    for (int i = 0; i < 32; ++i) dst[i * 512 + tid] = src[i * 512 + tid];
    kcolS[tid] = kcolG[tid];
  }

  {
    float* sd = (float*)W8;
    sd[tid] = p[tid];
    __syncthreads();
    for (int off = 256; off; off >>= 1) {
      if (tid < off) sd[tid] = fmaxf(sd[tid], sd[tid + off]);
      __syncthreads();
    }
    float M = sd[0];
    __syncthreads();
    sd[tid] = __expf(p[tid] - M);
    __syncthreads();
    for (int off = 256; off; off >>= 1) {
      if (tid < off) sd[tid] += sd[tid + off];
      __syncthreads();
    }
    if (tid == 0) *zpS = M + logf(sd[0]);
    __syncthreads();
  }
  const float Zp = *zpS;

  const int n0a = (wave * 2 + 0) * 16 + ln;
  const int n0b = (wave * 2 + 1) * 16 + ln;
  v16i bfrag[2][4];
  #pragma unroll
  for (int ti = 0; ti < 2; ++ti) {
    const size_t base = (size_t)(ti ? n0b : n0a) * S;
    #pragma unroll
    for (int kt = 0; kt < 4; ++kt) {
      #pragma unroll
      for (int d = 0; d < 16; ++d) {
        int kb = kt * 128 + (d >> 2) * 32 + laneHi * 16 + (d & 3) * 4;
        bfrag[ti][kt][d] = *(const int*)(T8 + base + kb);
      }
    }
  }
  const float kca = kcolS[n0a] + 8.0f;
  const float kcb = kcolS[n0b] + 8.0f;

  for (int b = 0; b < B; ++b)
    alpha[b * S + tid] = (p[tid] - Zp) + g[(long)(b * T) * S + tid];
  __syncthreads();

  for (int t = 1; t < T; ++t) {
    {
      const int b = wave;
      float m = -__builtin_inff();
      #pragma unroll
      for (int j = 0; j < 16; ++j) m = fmaxf(m, alpha[b * S + lane + 32 * j]);
      #pragma unroll
      for (int off = 16; off; off >>= 1) m = fmaxf(m, __shfl_xor(m, off));
      if (lane == 0) mrow[b] = m;
      #pragma unroll
      for (int j = 0; j < 16; ++j) {
        int s = lane + 32 * j;
        float w = exp2f(fmaf(alpha[b * S + s] - m, LOG2E, 8.0f));
        W8[b * S + s] = f32_to_fp8(w);
      }
    }
    __syncthreads();

    v8f acc0 = {}, acc1 = {};
    #pragma unroll
    for (int kt = 0; kt < 4; ++kt) {
      v16i a;
      #pragma unroll
      for (int d = 0; d < 16; ++d) {
        int h  = d >> 3, vv = d & 7;
        int ka = kt * 128 + (d >> 2) * 32 + laneHi * 16 + (d & 3) * 4;
        a[d] = *(const int*)(W8 + ln * S + ka);
      }
      acc0 = __builtin_amdgcn_wmma_f32_16x16x128_fp8_fp8(a, bfrag[0][kt], (short)0,
                                                         acc0, false, false);
      acc1 = __builtin_amdgcn_wmma_f32_16x16x128_fp8_fp8(a, bfrag[1][kt], (short)0,
                                                         acc1, false, false);
    }

    #pragma unroll
    for (int r = 0; r < 8; ++r) {
      int   m  = r + laneHi * 8;
      float e0 = g[(long)(m * T + t) * S + n0a];
      float e1 = g[(long)(m * T + t) * S + n0b];
      alpha[m * S + n0a] = mrow[m] + (__log2f(acc0[r]) - kca) * LN2 + e0;
      alpha[m * S + n0b] = mrow[m] + (__log2f(acc1[r]) - kcb) * LN2 + e1;
    }
    __syncthreads();
  }

  {
    const int b = wave;
    float m = -__builtin_inff();
    #pragma unroll
    for (int j = 0; j < 16; ++j) m = fmaxf(m, alpha[b * S + lane + 32 * j]);
    #pragma unroll
    for (int off = 16; off; off >>= 1) m = fmaxf(m, __shfl_xor(m, off));
    float s = 0.f;
    #pragma unroll
    for (int j = 0; j < 16; ++j)
      s += exp2f((alpha[b * S + lane + 32 * j] - m) * LOG2E);
    #pragma unroll
    for (int off = 16; off; off >>= 1) s += __shfl_xor(s, off);
    if (lane == 0) llrow[b] = m + __log2f(s) * LN2;
  }
  __syncthreads();
  if (tid == 0) {
    float acc = 0.f;
    for (int b = 0; b < B; ++b) acc += llrow[b];
    out[0] = -acc / (float)B;
  }
}

extern "C" void kernel_launch(void* const* d_in, const int* in_sizes, int n_in,
                              void* d_out, int out_size, void* d_ws, size_t ws_size,
                              hipStream_t stream) {
  const int*   ids = (const int*)d_in[0];
  const float* em  = (const float*)d_in[1];
  const float* tm  = (const float*)d_in[2];
  const float* p   = (const float*)d_in[3];
  float* out = (float*)d_out;

  char* ws = (char*)d_ws;
  float*         Zem  = (float*)ws;
  float*         Ztm  = (float*)(ws + 2048);
  float*         kcol = (float*)(ws + 4096);
  unsigned char* Ttr  = (unsigned char*)(ws + 8192);
  float*         g    = (float*)(ws + 8192 + 262144);

  hmm_zem   <<<S,     256, 0, stream>>>(em, Zem);
  hmm_ztm   <<<S,     256, 0, stream>>>(tm, Ztm);
  hmm_tcol  <<<S,     256, 0, stream>>>(tm, Ztm, Ttr, kcol);
  hmm_gather<<<B * T, 512, 0, stream>>>(ids, em, Zem, g);
  hmm_scan  <<<1,     512, 305408, stream>>>(p, Zem, Ttr, kcol, g, out);
}
